// TreeModel_75806172774616
// MI455X (gfx1250) — matrix operand order repaired and checked by reading, not compiled, not run on hardware
//
#include <hip/hip_runtime.h>
#include <math.h>

#define BETA_V   10.0f
#define BSZ      4096
#define DD       512
#define EE       512
#define NINNER   15
#define NLEAF    16

#if __has_builtin(__builtin_amdgcn_global_load_async_to_lds_b128)
#define HAVE_ASYNC_LDS 1
#else
#define HAVE_ASYNC_LDS 0
#endif

#define AS1 __attribute__((address_space(1)))
#define AS3 __attribute__((address_space(3)))

typedef int v4i __attribute__((vector_size(4 * sizeof(int))));

__device__ __forceinline__ void wait_asynccnt0() {
#if __has_builtin(__builtin_amdgcn_s_wait_asynccnt)
  __builtin_amdgcn_s_wait_asynccnt(0);
#else
  asm volatile("s_wait_asynccnt 0x0" ::: "memory");
#endif
}

typedef __attribute__((ext_vector_type(16))) __bf16 v16bf;
typedef __attribute__((ext_vector_type(8)))  float  v8f;

union Frag { uint4 q[2]; unsigned int u[8]; v16bf v; };

__device__ __forceinline__ unsigned int pack2_bf16(float x, float y) {
  unsigned int ux = __float_as_uint(x), uy = __float_as_uint(y);
  ux += 0x7FFFu + ((ux >> 16) & 1u);           // round-to-nearest-even
  uy += 0x7FFFu + ((uy >> 16) & 1u);
  return (ux >> 16) | (uy & 0xFFFF0000u);
}
__device__ __forceinline__ unsigned short to_bf16(float x) {
  unsigned int ux = __float_as_uint(x);
  ux += 0x7FFFu + ((ux >> 16) & 1u);
  return (unsigned short)(ux >> 16);
}
__device__ __forceinline__ float bf16_to_f32(unsigned short h) {
  return __uint_as_float(((unsigned int)h) << 16);
}

// ---------------------------------------------------------------------------
// Elementwise f32 -> bf16 (n multiple of 1024)
// ---------------------------------------------------------------------------
__global__ __launch_bounds__(256) void convert_bf16_kernel(
    const float* __restrict__ src, unsigned short* __restrict__ dst, long long n)
{
  long long i = ((long long)blockIdx.x * blockDim.x + threadIdx.x) * 4;
  if (i >= n) return;
  float4 f = *(const float4*)(src + i);
  uint2 o;
  o.x = pack2_bf16(f.x, f.y);
  o.y = pack2_bf16(f.z, f.w);
  *(uint2*)(dst + i) = o;
}

// ---------------------------------------------------------------------------
// Wt[z][n][k] (bf16) = W[z][k][n] (f32); 32x32 tiles via LDS.
// ---------------------------------------------------------------------------
__global__ __launch_bounds__(256) void transpose_convert_kernel(
    const float* __restrict__ W, unsigned short* __restrict__ Wt, int K, int N)
{
  __shared__ unsigned int lds[32 * 17];          // [k][n/2] bf16 pairs
  const int z  = blockIdx.z;
  const int k0 = blockIdx.x * 32;
  const int n0 = blockIdx.y * 32;
  const float* Wz = W + (long long)z * K * N;
  unsigned short* Wtz = Wt + (long long)z * K * N;
  const int t = threadIdx.x;
  {
    const int kr = t >> 3;
    const int nc = (t & 7) * 4;
    float4 f = *(const float4*)(Wz + (long long)(k0 + kr) * N + n0 + nc);
    lds[kr * 17 + (nc >> 1)]     = pack2_bf16(f.x, f.y);
    lds[kr * 17 + (nc >> 1) + 1] = pack2_bf16(f.z, f.w);
  }
  __syncthreads();
  {
    const int nr = t >> 3;
    const int kc = (t & 7) * 4;
    unsigned int u[4];
    #pragma unroll
    for (int i = 0; i < 4; ++i)
      u[i] = (lds[(kc + i) * 17 + (nr >> 1)] >> ((nr & 1) * 16)) & 0xFFFFu;
    uint2 o;
    o.x = u[0] | (u[1] << 16);
    o.y = u[2] | (u[3] << 16);
    *(uint2*)(Wtz + (long long)(n0 + nr) * K + k0 + kc) = o;
  }
}

// ---------------------------------------------------------------------------
// relu(A[z] @ Wt[z]^T + bias[z]) with M=4096, N=K=512, bf16 operands.
// A: [M][K] bf16 row-major; Wt: [N][K] bf16 (pre-transposed).
// Outputs: Cb bf16 (optional), Cf f32 (optional).
// Block tile 128x128, BK=32, 256 threads = 8 waves (2 in M x 4 in N);
// wave tile 64x32 = 4x2 v_wmma_f32_16x16x32_bf16 per K-step.
// Double-buffered LDS filled by GLOBAL_LOAD_ASYNC_TO_LDS_B128 (ASYNCcnt):
// tiles never touch VGPRs, DMA overlaps the WMMAs, one s_barrier per K-step.
// Fallback (no async builtin): register staging with stores before the WMMAs
// so the staging registers die before peak pressure (no scratch spill).
// ---------------------------------------------------------------------------
__global__ __launch_bounds__(256, 1) void gemm_bf16_bias_relu(
    const unsigned short* __restrict__ A, long long aStride,
    const unsigned short* __restrict__ Wt, long long wStride,
    const float* __restrict__ bias, long long bStride,
    unsigned short* __restrict__ Cb, long long cbStride,
    float* __restrict__ Cf, long long cfStride,
    int M, int N, int K)
{
  __shared__ uint4 ldsA4[2][128 * 5];   // row stride 5 uint4 (20 dwords)
  __shared__ uint4 ldsB4[2][128 * 5];

  const int tid   = threadIdx.x;
  const int lane  = tid & 31;
  const int wid   = tid >> 5;
  const int waveM = wid & 1;     // 64-row slab
  const int waveN = wid >> 1;    // 32-col slab
  const int z  = blockIdx.z;
  const int n0 = blockIdx.x * 128;
  const int m0 = blockIdx.y * 128;

  A    += (long long)z * aStride;
  Wt   += (long long)z * wStride;
  bias += (long long)z * bStride;

  const int half = lane >> 4;
  const int l15  = lane & 15;

  // staging: each thread owns 2 uint4 of A and 2 of B per K-step
  const int sRow = tid >> 1;           // 0..127
  const int sCol = (tid & 1) * 2;      // uint4 column {0,1} or {2,3}
  const unsigned short* aSrc = A  + (long long)(m0 + sRow) * K + sCol * 8;
  const unsigned short* bSrc = Wt + (long long)(n0 + sRow) * K + sCol * 8;
  const int sIdx = sRow * 5 + sCol;

#if HAVE_ASYNC_LDS
  // prologue: DMA tile 0 into lds[0]
  __builtin_amdgcn_global_load_async_to_lds_b128(
      (AS1 v4i*)(aSrc),     (AS3 v4i*)&ldsA4[0][sIdx],     0, 0);
  __builtin_amdgcn_global_load_async_to_lds_b128(
      (AS1 v4i*)(aSrc + 8), (AS3 v4i*)&ldsA4[0][sIdx + 1], 0, 0);
  __builtin_amdgcn_global_load_async_to_lds_b128(
      (AS1 v4i*)(bSrc),     (AS3 v4i*)&ldsB4[0][sIdx],     0, 0);
  __builtin_amdgcn_global_load_async_to_lds_b128(
      (AS1 v4i*)(bSrc + 8), (AS3 v4i*)&ldsB4[0][sIdx + 1], 0, 0);
#else
  {
    uint4 a0 = *(const uint4*)(aSrc);
    uint4 a1 = *(const uint4*)(aSrc + 8);
    uint4 b0 = *(const uint4*)(bSrc);
    uint4 b1 = *(const uint4*)(bSrc + 8);
    ldsA4[0][sIdx]     = a0;
    ldsA4[0][sIdx + 1] = a1;
    ldsB4[0][sIdx]     = b0;
    ldsB4[0][sIdx + 1] = b1;
  }
#endif

  v8f acc[4][2] = {};
  int buf = 0;

  for (int k0 = 0; k0 < K; k0 += 32) {
    const bool more = (k0 + 32 < K);

#if HAVE_ASYNC_LDS
    wait_asynccnt0();   // this wave's DMA into lds[buf] landed
    __syncthreads();    // all waves' DMA landed; prev readers of lds[buf^1] done
    if (more) {         // kick off DMA for tile k+1 into the other buffer
      const int nb = buf ^ 1;
      const unsigned short* an = aSrc + k0 + 32;
      const unsigned short* bn = bSrc + k0 + 32;
      __builtin_amdgcn_global_load_async_to_lds_b128(
          (AS1 v4i*)(an),     (AS3 v4i*)&ldsA4[nb][sIdx],     0, 0);
      __builtin_amdgcn_global_load_async_to_lds_b128(
          (AS1 v4i*)(an + 8), (AS3 v4i*)&ldsA4[nb][sIdx + 1], 0, 0);
      __builtin_amdgcn_global_load_async_to_lds_b128(
          (AS1 v4i*)(bn),     (AS3 v4i*)&ldsB4[nb][sIdx],     0, 0);
      __builtin_amdgcn_global_load_async_to_lds_b128(
          (AS1 v4i*)(bn + 8), (AS3 v4i*)&ldsB4[nb][sIdx + 1], 0, 0);
    }
#else
    uint4 aReg[2], bReg[2];
    if (more) {                          // issue next tile's global loads early
      const unsigned short* an = aSrc + k0 + 32;
      const unsigned short* bn = bSrc + k0 + 32;
      aReg[0] = *(const uint4*)(an);
      aReg[1] = *(const uint4*)(an + 8);
      bReg[0] = *(const uint4*)(bn);
      bReg[1] = *(const uint4*)(bn + 8);
      __builtin_prefetch(an + 32, 0, 1);
      __builtin_prefetch(bn + 32, 0, 1);
    }
    __syncthreads();
#endif

    // fragment loads (aligned b128) from lds[buf]
    Frag afrag[4], bfrag[2];
    #pragma unroll
    for (int i = 0; i < 4; ++i) {
      const int mrow = waveM * 64 + i * 16 + l15;
      // A 16x32 bf16 layout: k = (v>>2)*16 + half*8 + (v&3)*2
      afrag[i].q[0] = ldsA4[buf][mrow * 5 + half];
      afrag[i].q[1] = ldsA4[buf][mrow * 5 + 2 + half];
    }
    #pragma unroll
    for (int j = 0; j < 2; ++j) {
      const int ncol = waveN * 32 + j * 16 + l15;
      // B 32x16 bf16 layout: k = (v>>2)*16 + half*8 + (v&3)*2
      bfrag[j].q[0] = ldsB4[buf][ncol * 5 + half];
      bfrag[j].q[1] = ldsB4[buf][ncol * 5 + 2 + half];
    }

#if !HAVE_ASYNC_LDS
    if (more) {   // stage next tile now so aReg/bReg die before the WMMA block
      const int nb = buf ^ 1;
      ldsA4[nb][sIdx]     = aReg[0];
      ldsA4[nb][sIdx + 1] = aReg[1];
      ldsB4[nb][sIdx]     = bReg[0];
      ldsB4[nb][sIdx + 1] = bReg[1];
    }
#endif

    #pragma unroll
    for (int i = 0; i < 4; ++i)
      #pragma unroll
      for (int j = 0; j < 2; ++j)
        acc[i][j] = __builtin_amdgcn_wmma_f32_16x16x32_bf16(
            false, afrag[i].v, false, bfrag[j].v, (short)0, acc[i][j],
            false, false);

    if (more) buf ^= 1;
  }

  // epilogue: bias + relu; bf16 and/or f32 stores
  #pragma unroll
  for (int j = 0; j < 2; ++j) {
    const int n  = n0 + waveN * 32 + j * 16 + l15;
    const float bv = bias[n];
    #pragma unroll
    for (int i = 0; i < 4; ++i) {
      #pragma unroll
      for (int v = 0; v < 8; ++v) {
        const int m = m0 + waveM * 64 + i * 16 + half * 8 + v;
        float val = acc[i][j][v] + bv;
        val = val > 0.0f ? val : 0.0f;
        const long long idx = (long long)m * N + n;
        if (Cf) Cf[(long long)z * cfStride + idx] = val;
        if (Cb) Cb[(long long)z * cbStride + idx] = to_bf16(val);
      }
    }
  }
}

// ---------------------------------------------------------------------------
// Per (b,n): d* = argmax_d w[n,b,d] (first occurrence);
// sel = BETA*(x[b,d*] + tanh(h[n,b,:]·Wi3[n,:,d*] + bi3[n,d*]));
// p_right[b,n] = sigmoid(sel).
// (sparsemax pipeline collapses: w[idx] == 1 after the final renormalization)
// ---------------------------------------------------------------------------
__global__ __launch_bounds__(256) void node_pright_kernel(
    const float* __restrict__ w,              // [15][4096][512] f32
    const unsigned short* __restrict__ h,     // [15][4096][512] bf16
    const float* __restrict__ Wi3,            // [15][512][512]
    const float* __restrict__ bi3,            // [15][512]
    const float* __restrict__ x,              // [4096][512]
    float* __restrict__ p_right)              // [4096][15]
{
  const int b = blockIdx.x;
  const int n = blockIdx.y;
  const int tid = threadIdx.x;
  __shared__ float sval[256];
  __shared__ int   sidx[256];
  __shared__ float ssum[256];
  __shared__ int   sd;

  const float* wrow = w + ((long long)n * BSZ + b) * DD;
  float bestV = -1e30f; int bestI = 0;
  for (int d = tid; d < DD; d += 256) {
    const float v = wrow[d];
    if (v > bestV) { bestV = v; bestI = d; }
  }
  sval[tid] = bestV; sidx[tid] = bestI;
  __syncthreads();
  for (int s = 128; s > 0; s >>= 1) {
    if (tid < s) {
      const float v2 = sval[tid + s]; const int i2 = sidx[tid + s];
      if (v2 > sval[tid] || (v2 == sval[tid] && i2 < sidx[tid])) {
        sval[tid] = v2; sidx[tid] = i2;
      }
    }
    __syncthreads();
  }
  if (tid == 0) sd = sidx[0];
  __syncthreads();
  const int dstar = sd;

  const unsigned short* hrow = h + ((long long)n * BSZ + b) * EE;
  const float* wcol = Wi3 + (long long)n * EE * DD + dstar;
  float sum = 0.0f;
  for (int e = tid; e < EE; e += 256)
    sum += bf16_to_f32(hrow[e]) * wcol[(long long)e * DD];
  ssum[tid] = sum;
  __syncthreads();
  for (int s = 128; s > 0; s >>= 1) {
    if (tid < s) ssum[tid] += ssum[tid + s];
    __syncthreads();
  }
  if (tid == 0) {
    const float bv  = tanhf(ssum[0] + bi3[n * DD + dstar]);
    const float sel = BETA_V * (x[(long long)b * DD + dstar] + bv);
    p_right[b * NINNER + n] = 1.0f / (1.0f + expf(-sel));
  }
}

// ---------------------------------------------------------------------------
// leaf[b,n] = sigmoid(hl[n,b,:]·Wl2[n,:] + bl2[n])
// ---------------------------------------------------------------------------
__global__ __launch_bounds__(256) void leaf_kernel(
    const unsigned short* __restrict__ hl,    // [16][4096][512] bf16
    const float* __restrict__ Wl2,            // [16][512]
    const float* __restrict__ bl2,            // [16]
    float* __restrict__ leaf)                 // [4096][16]
{
  const int b = blockIdx.x, n = blockIdx.y, tid = threadIdx.x;
  __shared__ float ssum[256];
  const unsigned short* row = hl + ((long long)n * BSZ + b) * EE;
  const float* wv = Wl2 + n * EE;
  float sum = 0.0f;
  for (int e = tid; e < EE; e += 256) sum += bf16_to_f32(row[e]) * wv[e];
  ssum[tid] = sum;
  __syncthreads();
  for (int s = 128; s > 0; s >>= 1) {
    if (tid < s) ssum[tid] += ssum[tid + s];
    __syncthreads();
  }
  if (tid == 0)
    leaf[b * NLEAF + n] = 1.0f / (1.0f + expf(-(ssum[0] + bl2[n])));
}

// ---------------------------------------------------------------------------
// out[b] = sum_j leaf[b,j] * prod_path factors   (heap path = bits of j)
// ---------------------------------------------------------------------------
__global__ void route_kernel(const float* __restrict__ p_right,
                             const float* __restrict__ leaf,
                             float* __restrict__ out)
{
  const int b = blockIdx.x * blockDim.x + threadIdx.x;
  if (b >= BSZ) return;
  const float* p  = p_right + b * NINNER;
  const float* lf = leaf + b * NLEAF;
  float sum = 0.0f;
  #pragma unroll
  for (int j = 0; j < NLEAF; ++j) {
    int node = 0; float f = lf[j];
    #pragma unroll
    for (int t = 3; t >= 0; --t) {
      const int bit = (j >> t) & 1;
      const float pr = p[node];
      f *= bit ? pr : (1.0f - pr);
      node = 2 * node + 1 + bit;
    }
    sum += f;
  }
  out[b] = sum;
}

extern "C" void kernel_launch(void* const* d_in, const int* in_sizes, int n_in,
                              void* d_out, int out_size, void* d_ws, size_t ws_size,
                              hipStream_t stream)
{
  const float* x       = (const float*)d_in[0];
  const float* We0     = (const float*)d_in[1];
  const float* be0     = (const float*)d_in[2];
  const float* We_rest = (const float*)d_in[3];
  const float* be_rest = (const float*)d_in[4];
  const float* Wi1     = (const float*)d_in[5];
  const float* bi1     = (const float*)d_in[6];
  const float* Wi2     = (const float*)d_in[7];
  const float* bi2     = (const float*)d_in[8];
  const float* Wi3     = (const float*)d_in[9];
  const float* bi3     = (const float*)d_in[10];
  const float* Wl1     = (const float*)d_in[11];
  const float* bl1     = (const float*)d_in[12];
  const float* Wl2     = (const float*)d_in[13];
  const float* bl2     = (const float*)d_in[14];
  float* out = (float*)d_out;

  const size_t mat  = (size_t)EE * EE;                 // 262144 elems
  const size_t actB = (size_t)BSZ * EE;                // 2M elems
  char* ws = (char*)d_ws;
  size_t off = 0;
  unsigned short* xb   = (unsigned short*)(ws + off); off += actB * 2;           // 4 MiB
  unsigned short* wtb  = (unsigned short*)(ws + off); off += 50 * mat * 2;       // 25 MiB
  unsigned short* embA = (unsigned short*)(ws + off); off += actB * 2;
  unsigned short* embB = (unsigned short*)(ws + off); off += actB * 2;
  const size_t hwOff = off;
  unsigned short* hbuf = (unsigned short*)(ws + off); off += NINNER * actB * 2;  // 60 MiB
  float* wbuf          = (float*)(ws + off);          off += NINNER * actB * 4;  // 120 MiB
  float* pright        = (float*)(ws + off);          off += (size_t)BSZ * NINNER * 4;
  float* leafb         = (float*)(ws + off);          off += (size_t)BSZ * NLEAF * 4;
  unsigned short* hlbuf = (unsigned short*)(ws + hwOff);  // reuses h/w region (stream-ordered)

  unsigned short* We0t  = wtb;
  unsigned short* WerT  = wtb + 1 * mat;
  unsigned short* Wi1t  = wtb + 4 * mat;
  unsigned short* Wi2t  = wtb + 19 * mat;
  unsigned short* Wl1t  = wtb + 34 * mat;

  const long long WS = (long long)mat;     // weight batch stride (elems)
  const long long AS = (long long)actB;    // activation batch stride (elems)
  const dim3 blk(256);

  // ---- one-time per call: bf16 conversions (all L2-resident afterwards)
  convert_bf16_kernel<<<dim3((unsigned)(actB / 1024)), blk, 0, stream>>>(x, xb, (long long)actB);
  transpose_convert_kernel<<<dim3(EE/32, EE/32, 1),  blk, 0, stream>>>(We0,     We0t, EE, EE);
  transpose_convert_kernel<<<dim3(EE/32, EE/32, 3),  blk, 0, stream>>>(We_rest, WerT, EE, EE);
  transpose_convert_kernel<<<dim3(EE/32, EE/32, 15), blk, 0, stream>>>(Wi1,     Wi1t, EE, EE);
  transpose_convert_kernel<<<dim3(EE/32, EE/32, 15), blk, 0, stream>>>(Wi2,     Wi2t, EE, EE);
  transpose_convert_kernel<<<dim3(EE/32, EE/32, 16), blk, 0, stream>>>(Wl1,     Wl1t, EE, EE);

  const dim3 gemmGrid1(EE / 128, BSZ / 128, 1);
  // ---- embedding chain: xb -> embA -> embB -> embA -> embB  (bf16 out only)
  gemm_bf16_bias_relu<<<gemmGrid1, blk, 0, stream>>>(
      xb, 0, We0t, 0, be0, 0, embA, 0, (float*)nullptr, 0, BSZ, EE, DD);
  gemm_bf16_bias_relu<<<gemmGrid1, blk, 0, stream>>>(
      embA, 0, WerT + 0 * WS, 0, be_rest + 0 * EE, 0, embB, 0, (float*)nullptr, 0, BSZ, EE, EE);
  gemm_bf16_bias_relu<<<gemmGrid1, blk, 0, stream>>>(
      embB, 0, WerT + 1 * WS, 0, be_rest + 1 * EE, 0, embA, 0, (float*)nullptr, 0, BSZ, EE, EE);
  gemm_bf16_bias_relu<<<gemmGrid1, blk, 0, stream>>>(
      embA, 0, WerT + 2 * WS, 0, be_rest + 2 * EE, 0, embB, 0, (float*)nullptr, 0, BSZ, EE, EE);
  const unsigned short* emb = embB;

  // ---- h[n] = relu(emb @ Wi1[n] + bi1[n])  (bf16 out)
  const dim3 gemmGridI(EE / 128, BSZ / 128, NINNER);
  gemm_bf16_bias_relu<<<gemmGridI, blk, 0, stream>>>(
      emb, 0, Wi1t, WS, bi1, EE, hbuf, AS, (float*)nullptr, 0, BSZ, EE, EE);

  // ---- w[n] = relu(h[n] @ Wi2[n] + bi2[n])  (f32 out for argmax)
  gemm_bf16_bias_relu<<<gemmGridI, blk, 0, stream>>>(
      hbuf, AS, Wi2t, WS, bi2, DD, (unsigned short*)nullptr, 0, wbuf, AS, BSZ, DD, EE);

  // ---- p_right[b,n]
  node_pright_kernel<<<dim3(BSZ, NINNER), blk, 0, stream>>>(
      wbuf, hbuf, Wi3, bi3, x, pright);

  // ---- hl[n] = relu(emb @ Wl1[n] + bl1[n])  (bf16, reuses h/w scratch)
  const dim3 gemmGridL(EE / 128, BSZ / 128, NLEAF);
  gemm_bf16_bias_relu<<<gemmGridL, blk, 0, stream>>>(
      emb, 0, Wl1t, WS, bl1, EE, hlbuf, AS, (float*)nullptr, 0, BSZ, EE, EE);

  // ---- leaf[b,n]
  leaf_kernel<<<dim3(BSZ, NLEAF), blk, 0, stream>>>(hlbuf, Wl2, bl2, leafb);

  // ---- route + sum
  route_kernel<<<dim3(BSZ / 256), blk, 0, stream>>>(pright, leafb, out);

  (void)in_sizes; (void)n_in; (void)out_size; (void)ws_size;
}
